// DecoderFunc_81647328297612
// MI455X (gfx1250) — matrix operand order repaired and checked by reading, not compiled, not run on hardware
//
#include <hip/hip_runtime.h>
#include <hip/hip_bf16.h>

// ---------------- problem constants ----------------
#define NATOMS 64
#define KTYPES 2
#define VDIM   32
#define EDIM   32
#define HID    64
#define BSIMS  128
#define NEDGE  (NATOMS*(NATOMS-1))   // 4032
#define ROWS_E (BSIMS*NEDGE)         // 516096
#define ROWS_N (BSIMS*NATOMS)        // 8192
#define EPSBN  1e-5f

// ---------------- workspace layout (floats) ----------------
#define AGG_OFF    0
#define ONODE_OFF  (AGG_OFF + ROWS_N*EDIM)      // 262144
#define STATSN_OFF (ONODE_OFF + ROWS_N*VDIM)    // 524288 : sum[32], sumsq[32]
#define STATSE_OFF (STATSN_OFF + 64)            // 524352 : sum[2][32], sumsq[2][32]
#define MVN_OFF    (STATSE_OFF + 128)           // 524480 : mu[32], rstd[32]
#define MVE_OFF    (MVN_OFF + 64)               // 524544 : mu[2][32], rstd[2][32]
#define FRAG_OFF   (MVE_OFF + 128)              // 524672 (byte offset 32B-aligned)
#define NFRAGS      32
#define FRAG_HALVES 512                          // 32 lanes * 16 halves

typedef __attribute__((ext_vector_type(16))) _Float16 v16h;
typedef __attribute__((ext_vector_type(8)))  _Float16 v8h;
typedef __attribute__((ext_vector_type(8)))  float    v8f;

// ---------------- device helpers ----------------
__device__ __forceinline__ v8f wmma16(v16h a, v16h b, v8f c) {
  // D = A(16x32 f16) * B(32x16 f16) + C(f32)
  return __builtin_amdgcn_wmma_f32_16x16x32_f16(false, a, false, b, (short)0, c, false, false);
}

__device__ __forceinline__ v8f splat8(float x) {
  v8f r;
  #pragma unroll
  for (int i = 0; i < 8; ++i) r[i] = x;
  return r;
}

__device__ __forceinline__ float elu1(float x) { return x > 0.f ? x : (__expf(x) - 1.f); }

// Build an A fragment from f32 memory: two runs of 8 consecutive features
// (A layout: lane group g holds K = {8g..8g+7} in elems 0-7, {16+8g..23+8g} in elems 8-15)
__device__ __forceinline__ v16h a_from_f32(const float* run0, const float* run1) {
  float4 a0 = *(const float4*)run0, a1 = *(const float4*)(run0 + 4);
  float4 b0 = *(const float4*)run1, b1 = *(const float4*)(run1 + 4);
  v16h r;
  r[0]=(_Float16)a0.x; r[1]=(_Float16)a0.y; r[2]=(_Float16)a0.z; r[3]=(_Float16)a0.w;
  r[4]=(_Float16)a1.x; r[5]=(_Float16)a1.y; r[6]=(_Float16)a1.z; r[7]=(_Float16)a1.w;
  r[8]=(_Float16)b0.x; r[9]=(_Float16)b0.y; r[10]=(_Float16)b0.z; r[11]=(_Float16)b0.w;
  r[12]=(_Float16)b1.x; r[13]=(_Float16)b1.y; r[14]=(_Float16)b1.z; r[15]=(_Float16)b1.w;
  return r;
}

// A fragment from an f16 LDS row (runs are contiguous: [f0..f0+7], [f0+16..f0+23])
__device__ __forceinline__ v16h a_from_lds(const _Float16* row, int f0) {
  v8h lo = *(const v8h*)(row + f0);
  v8h hi = *(const v8h*)(row + f0 + 16);
  v16h r;
  #pragma unroll
  for (int i = 0; i < 8; ++i) { r[i] = lo[i]; r[8 + i] = hi[i]; }
  return r;
}

// Pre-packed B fragment: one 32B load per lane
__device__ __forceinline__ v16h ldfrag(const _Float16* wfrag, int id, int lane) {
  return *(const v16h*)(wfrag + id * FRAG_HALVES + lane * 16);
}

// ---------------- per-tile MLP cores ----------------
#define LDSW 72  // padded h-row stride (halves) to avoid LDS bank conflicts

// Edge MLP for one 16-row tile: features = [v0[recv](32) | v0[send](32)]
// frag ids: fc1 k*8 + nt*2 + kk (ids 0..15), fc2 16 + k*4 + t*2 + kk (ids 16..23)
__device__ __forceinline__ void edge_tile_compute(
    const float* __restrict__ v0, const _Float16* __restrict__ wfrag,
    const float* __restrict__ b1e, const float* __restrict__ b2e,
    int b, int e0i, _Float16* hl, int lane, float o[KTYPES][2][8])
{
  const int g = lane >> 4, m = lane & 15;
  const int e = e0i + m;
  const int si = e / (NATOMS - 1);
  const int jj = e - si * (NATOMS - 1);
  const int recv = jj + (jj >= si ? 1 : 0);
  const float* prec = v0 + ((size_t)b * NATOMS + recv) * VDIM;
  const float* psnd = v0 + ((size_t)b * NATOMS + si)   * VDIM;
  const v16h a0 = a_from_f32(prec + 8 * g, prec + 16 + 8 * g);  // kk=0 (recv feats)
  const v16h a1 = a_from_f32(psnd + 8 * g, psnd + 16 + 8 * g);  // kk=1 (send feats)
  _Float16* myrow = hl + m * LDSW;
  #pragma unroll
  for (int k = 0; k < KTYPES; ++k) {
    // fc1: 16x64 output = 4 tiles of 16x16, K=64 via two K=32 WMMAs
    #pragma unroll
    for (int nt = 0; nt < 4; ++nt) {
      v8f c = splat8(b1e[k * HID + nt * 16 + m]);
      c = wmma16(a0, ldfrag(wfrag, k * 8 + nt * 2 + 0, lane), c);
      c = wmma16(a1, ldfrag(wfrag, k * 8 + nt * 2 + 1, lane), c);
      #pragma unroll
      for (int v = 0; v < 8; ++v)  // C layout: row = v + 8g, col = nt*16 + m
        hl[(v + 8 * g) * LDSW + nt * 16 + m] = (_Float16)elu1(c[v]);
    }
    // transpose C-layout -> A-layout through LDS (in-order DS ops within wave)
    const v16h h0 = a_from_lds(myrow, 8 * g);
    const v16h h1 = a_from_lds(myrow, 32 + 8 * g);
    // fc2: 16x32 output = 2 tiles, K=64
    #pragma unroll
    for (int t = 0; t < 2; ++t) {
      v8f c = splat8(b2e[k * EDIM + t * 16 + m]);
      c = wmma16(h0, ldfrag(wfrag, 16 + k * 4 + t * 2 + 0, lane), c);
      c = wmma16(h1, ldfrag(wfrag, 16 + k * 4 + t * 2 + 1, lane), c);
      #pragma unroll
      for (int v = 0; v < 8; ++v) o[k][t][v] = elu1(c[v]);
    }
  }
}

// Node MLP for one 16-row tile: agg(32) -> 64 -> 32
// frag ids: fc1 24+nt (K=32), fc2 28 + t*2 + kk
__device__ __forceinline__ void node_tile_compute(
    const float* __restrict__ agg, const _Float16* __restrict__ wfrag,
    const float* __restrict__ b1v, const float* __restrict__ b2v,
    int r0, _Float16* hl, int lane, float o[2][8])
{
  const int g = lane >> 4, m = lane & 15;
  const float* pr = agg + (size_t)(r0 + m) * EDIM;
  const v16h a0 = a_from_f32(pr + 8 * g, pr + 16 + 8 * g);
  #pragma unroll
  for (int nt = 0; nt < 4; ++nt) {
    v8f c = splat8(b1v[nt * 16 + m]);
    c = wmma16(a0, ldfrag(wfrag, 24 + nt, lane), c);
    #pragma unroll
    for (int v = 0; v < 8; ++v)
      hl[(v + 8 * g) * LDSW + nt * 16 + m] = (_Float16)elu1(c[v]);
  }
  const _Float16* myrow = hl + m * LDSW;
  const v16h h0 = a_from_lds(myrow, 8 * g);
  const v16h h1 = a_from_lds(myrow, 32 + 8 * g);
  #pragma unroll
  for (int t = 0; t < 2; ++t) {
    v8f c = splat8(b2v[t * 16 + m]);
    c = wmma16(h0, ldfrag(wfrag, 28 + t * 2 + 0, lane), c);
    c = wmma16(h1, ldfrag(wfrag, 28 + t * 2 + 1, lane), c);
    #pragma unroll
    for (int v = 0; v < 8; ++v) o[t][v] = elu1(c[v]);
  }
}

// ---------------- kernels ----------------

// Pack f32 weights into f16 B-fragments in per-lane order.
// B layout: lane L holds col n = L%16; element i holds K = 8*(L/16) + i (i<8), 16 + 8*(L/16) + (i-8) (i>=8).
__global__ void pack_weights(const float* __restrict__ eW1, const float* __restrict__ eW2,
                             const float* __restrict__ vW1, const float* __restrict__ vW2,
                             _Float16* __restrict__ wfrag)
{
  const int id = blockIdx.x, lane = threadIdx.x;
  const int n = lane & 15, gk = lane >> 4;
  _Float16* dst = wfrag + id * FRAG_HALVES + lane * 16;
  #pragma unroll
  for (int i = 0; i < 16; ++i) {
    const int Kidx = (i < 8) ? (8 * gk + i) : (16 + 8 * gk + (i - 8));
    float v;
    if (id < 16) {            // edge fc1: W1_k [64][64]
      int k = id >> 3, rem = id & 7, nt = rem >> 1, kk = rem & 1;
      v = eW1[k * 64 * 64 + (kk * 32 + Kidx) * 64 + nt * 16 + n];
    } else if (id < 24) {     // edge fc2: W2_k [64][32]
      int t = id - 16, k = t >> 2, rem = t & 3, nt = rem >> 1, kk = rem & 1;
      v = eW2[k * 64 * 32 + (kk * 32 + Kidx) * 32 + nt * 16 + n];
    } else if (id < 28) {     // node fc1: [32][64]
      int nt = id - 24;
      v = vW1[Kidx * 64 + nt * 16 + n];
    } else {                  // node fc2: [64][32]
      int t = id - 28, nt = t >> 1, kk = t & 1;
      v = vW2[(kk * 32 + Kidx) * 32 + nt * 16 + n];
    }
    dst[i] = (_Float16)v;
  }
}

// edge2node scatter-add expressed as per-node gather: agg[b,j,c] = sum_{i!=j} e0[b,i,pos(j,i),c] / N
__global__ void agg_kernel(const float* __restrict__ e0, float* __restrict__ agg)
{
  const int idx = blockIdx.x * blockDim.x + threadIdx.x;  // B*N*32 threads
  const int c = idx & 31;
  const int bn = idx >> 5;
  const int j = bn & (NATOMS - 1);
  const int b = bn >> 6;
  const float* base = e0 + (size_t)b * NATOMS * (NATOMS - 1) * EDIM;
  float s = 0.f;
  for (int i = 0; i < NATOMS; ++i) {
    if (i == j) continue;
    const int pos = j - (j > i ? 1 : 0);
    s += base[((size_t)i * (NATOMS - 1) + pos) * EDIM + c];
  }
  agg[idx] = s * (1.0f / NATOMS);
}

#define NWAVES 4
#define NTPW   8
#define NBLOCKS 16   // 16*4*8 = 512 tiles = 8192 rows

__global__ __launch_bounds__(NWAVES * 32) void node_passA(
    const float* __restrict__ agg, const _Float16* __restrict__ wfrag,
    const float* __restrict__ b1v, const float* __restrict__ b2v,
    float* __restrict__ o_node, float* __restrict__ statsN)
{
  __shared__ _Float16 hls[NWAVES][16 * LDSW];
  const int wave = threadIdx.x >> 5, lane = threadIdx.x & 31;
  const int g = lane >> 4, m = lane & 15;
  const int gw = blockIdx.x * NWAVES + wave;
  float s1[2] = {0.f, 0.f}, s2[2] = {0.f, 0.f};
  for (int it = 0; it < NTPW; ++it) {
    const int r0 = (gw * NTPW + it) * 16;
    float o[2][8];
    node_tile_compute(agg, wfrag, b1v, b2v, r0, hls[wave], lane, o);
    #pragma unroll
    for (int t = 0; t < 2; ++t)
      #pragma unroll
      for (int v = 0; v < 8; ++v) {
        o_node[((size_t)r0 + v + 8 * g) * VDIM + t * 16 + m] = o[t][v];
        s1[t] += o[t][v]; s2[t] += o[t][v] * o[t][v];
      }
  }
  #pragma unroll
  for (int t = 0; t < 2; ++t) { s1[t] += __shfl_xor(s1[t], 16); s2[t] += __shfl_xor(s2[t], 16); }
  if (lane < 16) {
    #pragma unroll
    for (int t = 0; t < 2; ++t) {
      atomicAdd(&statsN[t * 16 + lane], s1[t]);
      atomicAdd(&statsN[VDIM + t * 16 + lane], s2[t]);
    }
  }
}

#define EWAVES 4
#define ETPW   16
#define EBLOCKS 504  // 504*4*16 = 32256 tiles = 516096 rows

__global__ __launch_bounds__(EWAVES * 32) void edge_passA(
    const float* __restrict__ v0, const _Float16* __restrict__ wfrag,
    const float* __restrict__ b1e, const float* __restrict__ b2e,
    float* __restrict__ statsE)
{
  __shared__ _Float16 hls[EWAVES][16 * LDSW];
  const int wave = threadIdx.x >> 5, lane = threadIdx.x & 31;
  const int gw = blockIdx.x * EWAVES + wave;
  float s1[KTYPES][2] = {{0.f}}, s2[KTYPES][2] = {{0.f}};
  for (int it = 0; it < ETPW; ++it) {
    const int tile = gw * ETPW + it;
    const int r0 = tile * 16;
    const int b = r0 / NEDGE;
    const int e0i = r0 - b * NEDGE;
    float o[KTYPES][2][8];
    edge_tile_compute(v0, wfrag, b1e, b2e, b, e0i, hls[wave], lane, o);
    #pragma unroll
    for (int k = 0; k < KTYPES; ++k)
      #pragma unroll
      for (int t = 0; t < 2; ++t)
        #pragma unroll
        for (int v = 0; v < 8; ++v) { float x = o[k][t][v]; s1[k][t] += x; s2[k][t] += x * x; }
  }
  #pragma unroll
  for (int k = 0; k < KTYPES; ++k)
    #pragma unroll
    for (int t = 0; t < 2; ++t) {
      s1[k][t] += __shfl_xor(s1[k][t], 16);
      s2[k][t] += __shfl_xor(s2[k][t], 16);
    }
  if (lane < 16) {
    #pragma unroll
    for (int k = 0; k < KTYPES; ++k)
      #pragma unroll
      for (int t = 0; t < 2; ++t) {
        atomicAdd(&statsE[k * EDIM + t * 16 + lane], s1[k][t]);
        atomicAdd(&statsE[KTYPES * EDIM + k * EDIM + t * 16 + lane], s2[k][t]);
      }
  }
}

__global__ void finalize_stats(const float* __restrict__ statsN, const float* __restrict__ statsE,
                               float* __restrict__ mvN, float* __restrict__ mvE)
{
  const int t = threadIdx.x;
  if (t < VDIM) {
    const float inv = 1.0f / (float)ROWS_N;
    const float mu = statsN[t] * inv;
    const float var = statsN[VDIM + t] * inv - mu * mu;
    mvN[t] = mu; mvN[VDIM + t] = rsqrtf(var + EPSBN);
  } else if (t < VDIM + KTYPES * EDIM) {
    const int c = t - VDIM;
    const float inv = 1.0f / (float)ROWS_E;
    const float mu = statsE[c] * inv;
    const float var = statsE[KTYPES * EDIM + c] * inv - mu * mu;
    mvE[c] = mu; mvE[KTYPES * EDIM + c] = rsqrtf(var + EPSBN);
  }
}

__global__ void node_norm(const float* __restrict__ onode, const float* __restrict__ mvN,
                          const float* __restrict__ gamma, const float* __restrict__ beta,
                          float* __restrict__ dv)
{
  const int idx = blockIdx.x * blockDim.x + threadIdx.x;
  const int c = idx & (VDIM - 1);
  const float o = onode[idx];
  dv[idx] = gamma[c] * (o - mvN[c]) * mvN[VDIM + c] + beta[c];
}

__global__ __launch_bounds__(EWAVES * 32) void edge_passB(
    const float* __restrict__ v0, const float* __restrict__ etype,
    const _Float16* __restrict__ wfrag,
    const float* __restrict__ b1e, const float* __restrict__ b2e,
    const float* __restrict__ gammaE, const float* __restrict__ betaE,
    const float* __restrict__ mvE, float* __restrict__ de)
{
  __shared__ _Float16 hls[EWAVES][16 * LDSW];
  const int wave = threadIdx.x >> 5, lane = threadIdx.x & 31;
  const int g = lane >> 4, m = lane & 15;
  const int gw = blockIdx.x * EWAVES + wave;
  for (int it = 0; it < ETPW; ++it) {
    const int tile = gw * ETPW + it;
    const int r0 = tile * 16;
    const int b = r0 / NEDGE;
    const int e0i = r0 - b * NEDGE;
    float o[KTYPES][2][8];
    edge_tile_compute(v0, wfrag, b1e, b2e, b, e0i, hls[wave], lane, o);
    float et[8][2];
    #pragma unroll
    for (int v = 0; v < 8; ++v) {
      const size_t r = (size_t)r0 + v + 8 * g;
      const float2 q = *(const float2*)(etype + r * KTYPES);
      et[v][0] = q.x; et[v][1] = q.y;
    }
    float acc[2][8] = {{0.f}};
    #pragma unroll
    for (int k = 0; k < KTYPES; ++k)
      #pragma unroll
      for (int t = 0; t < 2; ++t) {
        const int c = k * EDIM + t * 16 + m;
        const float mu = mvE[c], rs = mvE[KTYPES * EDIM + c];
        const float ga = gammaE[c], be = betaE[c];
        #pragma unroll
        for (int v = 0; v < 8; ++v)
          acc[t][v] += (ga * (o[k][t][v] - mu) * rs + be) * et[v][k];
      }
    #pragma unroll
    for (int t = 0; t < 2; ++t)
      #pragma unroll
      for (int v = 0; v < 8; ++v)
        de[((size_t)r0 + v + 8 * g) * EDIM + t * 16 + m] = acc[t][v];
  }
}

// ---------------- host launcher ----------------
extern "C" void kernel_launch(void* const* d_in, const int* in_sizes, int n_in,
                              void* d_out, int out_size, void* d_ws, size_t ws_size,
                              hipStream_t stream)
{
  const float* v0    = (const float*)d_in[0];
  const float* e0    = (const float*)d_in[1];
  const float* etype = (const float*)d_in[2];
  const float* vW1   = (const float*)d_in[3];
  const float* vb1   = (const float*)d_in[4];
  const float* vW2   = (const float*)d_in[5];
  const float* vb2   = (const float*)d_in[6];
  const float* vga   = (const float*)d_in[7];
  const float* vbe   = (const float*)d_in[8];
  const float* eW1   = (const float*)d_in[9];
  const float* eb1   = (const float*)d_in[10];
  const float* eW2   = (const float*)d_in[11];
  const float* eb2   = (const float*)d_in[12];
  const float* ega   = (const float*)d_in[13];
  const float* ebe   = (const float*)d_in[14];

  float* out = (float*)d_out;
  float* wsf = (float*)d_ws;
  float* agg    = wsf + AGG_OFF;
  float* onode  = wsf + ONODE_OFF;
  float* statsN = wsf + STATSN_OFF;
  float* statsE = wsf + STATSE_OFF;
  float* mvN    = wsf + MVN_OFF;
  float* mvE    = wsf + MVE_OFF;
  _Float16* wfrag = (_Float16*)(wsf + FRAG_OFF);

  float* dv = out;                                   // [B,N,32]
  float* de = out + (size_t)ROWS_N * VDIM;           // [B,E,32]
  float* z  = de + (size_t)ROWS_E * EDIM;            // [B,E,K] -> zeros

  // stats must be zeroed every call (graph replay-safe: memset nodes)
  hipMemsetAsync(statsN, 0, (64 + 128) * sizeof(float), stream);
  hipMemsetAsync(z, 0, (size_t)ROWS_E * KTYPES * sizeof(float), stream);

  pack_weights<<<NFRAGS, 32, 0, stream>>>(eW1, eW2, vW1, vW2, wfrag);
  agg_kernel<<<(ROWS_N * EDIM) / 256, 256, 0, stream>>>(e0, agg);
  node_passA<<<NBLOCKS, NWAVES * 32, 0, stream>>>(agg, wfrag, vb1, vb2, onode, statsN);
  edge_passA<<<EBLOCKS, EWAVES * 32, 0, stream>>>(v0, wfrag, eb1, eb2, statsE);
  finalize_stats<<<1, 96, 0, stream>>>(statsN, statsE, mvN, mvE);
  node_norm<<<(ROWS_N * VDIM) / 256, 256, 0, stream>>>(onode, mvN, vga, vbe, dv);
  edge_passB<<<EBLOCKS, EWAVES * 32, 0, stream>>>(v0, etype, wfrag, eb1, eb2, ega, ebe, mvE, de);
}
